// TriangularEdgeUpdate_55877524521601
// MI455X (gfx1250) — compile-verified, matrix operand order checked by reading, not run on hardware
//
#include <hip/hip_runtime.h>
#include <hip/hip_bf16.h>

// ---------------------------------------------------------------------------
// TriangularEdgeUpdate for MI455X (gfx1250, wave32, WMMA bf16 16x16x32)
//   B=1, N=512, C_IN=C_CH=128
// Pipeline:
//   kW : f32 [in][out] weights -> bf16 [out][in]   (6 matrices)
//   kA : LN(z) -> 5 projections (WMMA) -> gating/mask -> planar bf16 aT/bT/gT
//   kB : k[i,j,o] = sum_k aT[o][i][k]*bT[o][j][k]  (128 batched 512^3 GEMMs)
//   kC : LN_o(k) @ Wo + bo, * gate   (WMMA, in-place on d_out)
// ---------------------------------------------------------------------------

#define NN 512
#define CC 128

typedef __attribute__((ext_vector_type(16))) __bf16 v16bf;
typedef __attribute__((ext_vector_type(8)))  __bf16 v8bf;
typedef __attribute__((ext_vector_type(8)))  float  v8f;
typedef __attribute__((ext_vector_type(4)))  float  v4f;
typedef __attribute__((ext_vector_type(2)))  float  v2f;

union Frag16 { v16bf v; v8bf h[2]; };

// A/B fragment (16x32 bf16): lane L -> row = L%16, two contiguous 8-elem runs
// at k = koff and k = koff+16, with koff = chunkBase + 8*(L/16).
__device__ __forceinline__ v16bf load_frag(const __bf16* __restrict__ row, int koff) {
    Frag16 f;
    f.h[0] = *(const v8bf*)(row + koff);
    f.h[1] = *(const v8bf*)(row + koff + 16);
    return f.v;
}

__device__ __forceinline__ float sigmoidf(float x) {
    return 1.0f / (1.0f + __expf(-x));
}

// ---------------------------------------------------------------------------
// kW: convert+transpose weights. W[in][out] f32 -> Wt[out][in] bf16.
// ---------------------------------------------------------------------------
__global__ __launch_bounds__(256) void kW(
    const float* __restrict__ Wa,  const float* __restrict__ Wga,
    const float* __restrict__ Wb,  const float* __restrict__ Wgb,
    const float* __restrict__ Wgo, const float* __restrict__ Wo,
    __bf16* __restrict__ Wt)
{
    const float* src[6] = {Wa, Wga, Wb, Wgb, Wgo, Wo};
    const float* W = src[blockIdx.x];
    __bf16* dst = Wt + (long)blockIdx.x * CC * CC;
    for (int idx = threadIdx.x; idx < CC * CC; idx += 256) {
        int o = idx >> 7, i = idx & 127;     // dst[o][i] = W[i][o]
        dst[idx] = (__bf16)W[i * CC + o];
    }
}

// ---------------------------------------------------------------------------
// kA: per 16-pixel tile: LayerNorm, 5 WMMA matmuls zn[16x128]@W[128x128],
// gating, masking, channel-planar transposed stores (packed 16B).
// Block = 128 threads = 4 waves; wave w handles channel tiles 2w, 2w+1.
// ---------------------------------------------------------------------------
__global__ __launch_bounds__(128) void kA(
    const float* __restrict__ z, const unsigned char* __restrict__ mask,
    const float* __restrict__ g_in, const float* __restrict__ b_in,
    const float* __restrict__ ba,  const float* __restrict__ bga,
    const float* __restrict__ bb,  const float* __restrict__ bgb,
    const float* __restrict__ bgo,
    const __bf16* __restrict__ Wt,          // [5][128][128] bf16 (out-major)
    __bf16* __restrict__ aT, __bf16* __restrict__ bT,
    __bf16* __restrict__ gT)                // gate, planar [ch][p] bf16
{
    __shared__ __align__(16) float  sZ[16][CC];
    __shared__ float sMu[16], sRs[16];
    __shared__ __align__(16) __bf16 sZn[16][CC];

    const int  t  = threadIdx.x;
    const long p0 = (long)blockIdx.x * 16;

    // ---- load z tile (16 x 128 f32) ----
    for (int idx = t; idx < 16 * CC; idx += 128)
        sZ[idx >> 7][idx & 127] = z[p0 * CC + idx];
    __syncthreads();

    // ---- layernorm stats: 8 threads per pixel + shfl_xor tree ----
    {
        const int p = t >> 3, c0 = (t & 7) * 16;
        float s = 0.f, s2 = 0.f;
#pragma unroll
        for (int c = 0; c < 16; ++c) { float v = sZ[p][c0 + c]; s += v; s2 += v * v; }
        s  += __shfl_xor(s, 1);  s2 += __shfl_xor(s2, 1);
        s  += __shfl_xor(s, 2);  s2 += __shfl_xor(s2, 2);
        s  += __shfl_xor(s, 4);  s2 += __shfl_xor(s2, 4);
        if ((t & 7) == 0) {
            float mu  = s  * (1.0f / CC);
            float var = s2 * (1.0f / CC) - mu * mu;
            sMu[p] = mu;
            sRs[p] = rsqrtf(var + 1e-5f);
        }
    }
    __syncthreads();

    // ---- normalize -> bf16 ----
    for (int idx = t; idx < 16 * CC; idx += 128) {
        int r = idx >> 7, c = idx & 127;
        sZn[r][c] = (__bf16)((sZ[r][c] - sMu[r]) * sRs[r] * g_in[c] + b_in[c]);
    }
    __syncthreads();

    const int lane = t & 31, wave = t >> 5;
    const int row  = lane & 15, half = lane >> 4;

    // A fragments: zn is the 16x128 A matrix, 4 K-chunks of 32
    v16bf af[4];
#pragma unroll
    for (int q = 0; q < 4; ++q)
        af[q] = load_frag(&sZn[row][0], q * 32 + 8 * half);

#pragma unroll
    for (int tt = 0; tt < 2; ++tt) {
        const int ct  = wave * 2 + tt;      // channel tile 0..7
        const int ch0 = ct * 16;
        v8f acc[5] = {};
#pragma unroll
        for (int m = 0; m < 5; ++m) {
            const __bf16* wrow = Wt + ((long)m * CC + ch0 + row) * CC;
#pragma unroll
            for (int q = 0; q < 4; ++q) {
                v16bf bf = load_frag(wrow, q * 32 + 8 * half);
                acc[m] = __builtin_amdgcn_wmma_f32_16x16x32_bf16(
                    false, af[q], false, bf, (short)0, acc[m], false, false);
            }
        }

        // epilogue: this lane's output channel n = ch0 + row (C-frag N index)
        const int   n     = ch0 + row;
        const float ba_n  = ba[n],  bga_n = bga[n];
        const float bb_n  = bb[n],  bgb_n = bgb[n];
        const float bgo_n = bgo[n];

        v8bf av, bv, gv;
#pragma unroll
        for (int v = 0; v < 8; ++v) {
            const int  Mv = v + 8 * half;
            const long p  = p0 + Mv;
            const float mz = mask[p] ? 0.0f : 1.0f;   // where(mask, 0, x)
            float aval = sigmoidf(acc[1][v] + bga_n) * (acc[0][v] + ba_n) * mz;
            float bval = sigmoidf(acc[3][v] + bgb_n) * (acc[2][v] + bb_n) * mz;
            av[v] = (__bf16)aval;
            bv[v] = (__bf16)bval;
            gv[v] = (__bf16)sigmoidf(acc[4][v] + bgo_n);
        }
        // planar store: 8 consecutive pixels -> one 16B store each
        const long off = (long)n * (NN * NN) + p0 + 8 * half;
        *(v8bf*)(aT + off) = av;
        *(v8bf*)(bT + off) = bv;
        *(v8bf*)(gT + off) = gv;
    }
}

// ---------------------------------------------------------------------------
// kB: batched per-channel GEMM  k[i,j,o] = A_o[i,:] . B_o[j,:]
// Grid (16,16,8): 32x32 (i,j) tile x 16-channel group. Block = 256 thr = 8 waves.
// Wave w owns channels ch0w = 16*bz + 2w .. +1 and a 32x32 register tile:
//   8 chains (2ch x 2isub x 2jsub) sharing 4 A-frags + 4 B-frags per K-chunk
//   -> 16 b128 loads per 8 WMMAs (2:1), zero in-block load redundancy.
// Adjacent-channel pairs allow packed float2 stores to [i][j][o].
// ---------------------------------------------------------------------------
__global__ __launch_bounds__(256) void kB(
    const __bf16* __restrict__ aT, const __bf16* __restrict__ bT,
    float* __restrict__ kout)
{
    const int t = threadIdx.x, lane = t & 31, wave = t >> 5;
    const int row = lane & 15, half = lane >> 4;
    const int i0   = blockIdx.x * 32;
    const int j0   = blockIdx.y * 32;
    const int ch0w = blockIdx.z * 16 + wave * 2;   // this wave's channel pair

    const __bf16* ar[2][2];   // [ch2][isub]
    const __bf16* br[2][2];   // [ch2][jsub]
#pragma unroll
    for (int c2 = 0; c2 < 2; ++c2) {
        const long plane = (long)(ch0w + c2) * (NN * NN);
#pragma unroll
        for (int s = 0; s < 2; ++s) {
            ar[c2][s] = aT + plane + (long)(i0 + s * 16 + row) * NN;
            br[c2][s] = bT + plane + (long)(j0 + s * 16 + row) * NN;
        }
    }

    v8f acc[2][2][2] = {};    // [ch2][isub][jsub]
    for (int kc = 0; kc < NN; kc += 32) {
        const int koff = kc + 8 * half;
        if ((kc & 63) == 0 && kc + 256 < NN) {
#pragma unroll
            for (int c2 = 0; c2 < 2; ++c2)
#pragma unroll
                for (int s = 0; s < 2; ++s) {
                    __builtin_prefetch(ar[c2][s] + kc + 256, 0, 3);
                    __builtin_prefetch(br[c2][s] + kc + 256, 0, 3);
                }
        }
#pragma unroll
        for (int c2 = 0; c2 < 2; ++c2) {
            v16bf afr0 = load_frag(ar[c2][0], koff);
            v16bf afr1 = load_frag(ar[c2][1], koff);
            v16bf bfr0 = load_frag(br[c2][0], koff);
            v16bf bfr1 = load_frag(br[c2][1], koff);
            acc[c2][0][0] = __builtin_amdgcn_wmma_f32_16x16x32_bf16(
                false, afr0, false, bfr0, (short)0, acc[c2][0][0], false, false);
            acc[c2][0][1] = __builtin_amdgcn_wmma_f32_16x16x32_bf16(
                false, afr0, false, bfr1, (short)0, acc[c2][0][1], false, false);
            acc[c2][1][0] = __builtin_amdgcn_wmma_f32_16x16x32_bf16(
                false, afr1, false, bfr0, (short)0, acc[c2][1][0], false, false);
            acc[c2][1][1] = __builtin_amdgcn_wmma_f32_16x16x32_bf16(
                false, afr1, false, bfr1, (short)0, acc[c2][1][1], false, false);
        }
    }

    // packed float2 stores: channel pair is contiguous in [i][j][o]
#pragma unroll
    for (int is = 0; is < 2; ++is)
#pragma unroll
        for (int js = 0; js < 2; ++js) {
            const int j = j0 + js * 16 + row;          // C-frag N
#pragma unroll
            for (int v = 0; v < 8; ++v) {
                const int i = i0 + is * 16 + v + 8 * half;  // C-frag M
                v2f pr;
                pr.x = acc[0][is][js][v];
                pr.y = acc[1][is][js][v];
                *(v2f*)(kout + ((long)i * NN + j) * CC + ch0w) = pr;
            }
        }
}

// ---------------------------------------------------------------------------
// kC: out[p][:] = gate[p][:] * (LN_o(k[p][:]) @ Wo + bo), in place on d_out.
// Output staged via LDS for coalesced float4 stores.
// ---------------------------------------------------------------------------
__global__ __launch_bounds__(128) void kC(
    float* __restrict__ kio,                 // d_out, holds k, rewritten
    const float* __restrict__ g_o, const float* __restrict__ b_o,
    const float* __restrict__ bo,
    const __bf16* __restrict__ WoT,          // [128 out][128 in] bf16
    const __bf16* __restrict__ gT)           // gate planar [ch][p] bf16
{
    __shared__ __align__(16) float  sK[16][CC];   // k tile, reused as out stage
    __shared__ float sMu[16], sRs[16];
    __shared__ __align__(16) __bf16 sKn[16][CC];

    const int  t  = threadIdx.x;
    const long p0 = (long)blockIdx.x * 16;

    for (int idx = t; idx < 16 * CC; idx += 128)
        sK[idx >> 7][idx & 127] = kio[p0 * CC + idx];
    __syncthreads();

    {
        const int p = t >> 3, c0 = (t & 7) * 16;
        float s = 0.f, s2 = 0.f;
#pragma unroll
        for (int c = 0; c < 16; ++c) { float v = sK[p][c0 + c]; s += v; s2 += v * v; }
        s  += __shfl_xor(s, 1);  s2 += __shfl_xor(s2, 1);
        s  += __shfl_xor(s, 2);  s2 += __shfl_xor(s2, 2);
        s  += __shfl_xor(s, 4);  s2 += __shfl_xor(s2, 4);
        if ((t & 7) == 0) {
            float mu  = s  * (1.0f / CC);
            float var = s2 * (1.0f / CC) - mu * mu;
            sMu[p] = mu;
            sRs[p] = rsqrtf(var + 1e-5f);
        }
    }
    __syncthreads();

    for (int idx = t; idx < 16 * CC; idx += 128) {
        int r = idx >> 7, c = idx & 127;
        sKn[r][c] = (__bf16)((sK[r][c] - sMu[r]) * sRs[r] * g_o[c] + b_o[c]);
    }
    __syncthreads();

    const int lane = t & 31, wave = t >> 5;
    const int row  = lane & 15, half = lane >> 4;

    v16bf af[4];
#pragma unroll
    for (int q = 0; q < 4; ++q)
        af[q] = load_frag(&sKn[row][0], q * 32 + 8 * half);

#pragma unroll
    for (int tt = 0; tt < 2; ++tt) {
        const int ct  = wave * 2 + tt;
        const int ch0 = ct * 16;
        v8f acc = {};
        const __bf16* wrow = WoT + (long)(ch0 + row) * CC;
#pragma unroll
        for (int q = 0; q < 4; ++q) {
            v16bf bf = load_frag(wrow, q * 32 + 8 * half);
            acc = __builtin_amdgcn_wmma_f32_16x16x32_bf16(
                false, af[q], false, bf, (short)0, acc, false, false);
        }
        const int   n    = ch0 + row;
        const float bo_n = bo[n];
        // vector gate load: 8 consecutive pixels from planar gate
        const v8bf gvec = *(const v8bf*)(gT + (long)n * (NN * NN) + p0 + 8 * half);
#pragma unroll
        for (int v = 0; v < 8; ++v) {
            const int Mv = v + 8 * half;
            sK[Mv][n] = (float)gvec[v] * (acc[v] + bo_n);   // stage into LDS
        }
    }
    __syncthreads();

    // coalesced float4 write-back
    const v4f* sOut4 = (const v4f*)&sK[0][0];
    for (int idx = t; idx < (16 * CC) / 4; idx += 128)
        *(v4f*)(kio + p0 * CC + idx * 4) = sOut4[idx];
}

// ---------------------------------------------------------------------------
// Launch
// ---------------------------------------------------------------------------
extern "C" void kernel_launch(void* const* d_in, const int* in_sizes, int n_in,
                              void* d_out, int out_size, void* d_ws, size_t ws_size,
                              hipStream_t stream) {
    const float*         z    = (const float*)d_in[0];
    const unsigned char* mask = (const unsigned char*)d_in[1];
    const float* g_in = (const float*)d_in[2];
    const float* b_in = (const float*)d_in[3];
    const float* Wa   = (const float*)d_in[4];
    const float* ba   = (const float*)d_in[5];
    const float* Wga  = (const float*)d_in[6];
    const float* bga  = (const float*)d_in[7];
    const float* Wb   = (const float*)d_in[8];
    const float* bb   = (const float*)d_in[9];
    const float* Wgb  = (const float*)d_in[10];
    const float* bgb  = (const float*)d_in[11];
    const float* g_o  = (const float*)d_in[12];
    const float* b_o  = (const float*)d_in[13];
    const float* Wgo  = (const float*)d_in[14];
    const float* bgo  = (const float*)d_in[15];
    const float* Wo   = (const float*)d_in[16];
    const float* bo   = (const float*)d_in[17];
    float* out = (float*)d_out;

    // workspace map
    const size_t WT_OFF = 0;                                   // 192KB
    const size_t AT_OFF = 1u << 20;                            // 64 MB
    const size_t BT_OFF = AT_OFF + 67108864ull;                // 64 MB
    const size_t GT_OFF = BT_OFF + 67108864ull;                // 64 MB
    char* ws = (char*)d_ws;
    __bf16* Wt = (__bf16*)(ws + WT_OFF);
    __bf16* aT = (__bf16*)(ws + AT_OFF);
    __bf16* bT = (__bf16*)(ws + BT_OFF);
    __bf16* gT = (__bf16*)(ws + GT_OFF);

    kW<<<6, 256, 0, stream>>>(Wa, Wga, Wb, Wgb, Wgo, Wo, Wt);

    kA<<<(NN * NN) / 16, 128, 0, stream>>>(
        z, mask, g_in, b_in, ba, bga, bb, bgb, bgo, Wt, aT, bT, gT);

    kB<<<dim3(NN / 32, NN / 32, CC / 16), 256, 0, stream>>>(aT, bT, out);

    kC<<<(NN * NN) / 16, 128, 0, stream>>>(
        out, g_o, b_o, bo, Wt + 5 * CC * CC, gT);
}
